// CompressiveMemory_39367670235839
// MI455X (gfx1250) — matrix operand order repaired and checked by reading, not compiled, not run on hardware
//
#include <hip/hip_runtime.h>
#include <hip/hip_bf16.h>

typedef __attribute__((ext_vector_type(16))) _Float16 v16h;
typedef __attribute__((ext_vector_type(8)))  _Float16 v8h;
typedef __attribute__((ext_vector_type(8)))  float    v8f;
typedef __attribute__((ext_vector_type(4)))  float    v4f;
typedef __attribute__((ext_vector_type(4)))  int      v4i;

#define AS1 __attribute__((address_space(1)))
#define AS3 __attribute__((address_space(3)))

#if __has_builtin(__builtin_amdgcn_global_load_async_to_lds_b128) && \
    __has_builtin(__builtin_amdgcn_s_wait_asynccnt)
#define HAVE_ASYNC_LDS 1
#else
#define HAVE_ASYNC_LDS 0
#endif

// ---------------------------------------------------------------------------
// WMMA helpers (CDNA5 v_wmma_f32_16x16x32_f16)
// ---------------------------------------------------------------------------
__device__ __forceinline__ v8f wmma_f16(v16h a, v16h b, v8f c) {
  return __builtin_amdgcn_wmma_f32_16x16x32_f16(false, a, false, b, (short)0, c,
                                                false, false);
}

// A-fragment (16x32 f16, row-major [M][K]); rowPtr already includes (lane&15) row.
// lanes 0-15: elems 0-7 = K[0..7],  8-15 = K[16..23]
// lanes16-31: elems 0-7 = K[8..15], 8-15 = K[24..31]
__device__ __forceinline__ v16h load_fragA(const _Float16* rowPtr, int kOff, int lane) {
  const int h = (lane >> 4) & 1;
  v8h lo = *(const v8h*)(rowPtr + kOff + h * 8);
  v8h hi = *(const v8h*)(rowPtr + kOff + 16 + h * 8);
  v16h f;
#pragma unroll
  for (int i = 0; i < 8; ++i) { f[i] = lo[i]; f[i + 8] = hi[i]; }
  return f;
}

// B-fragment (32x16 f16): lane L holds column N=L&15, K order as A-fragment.
// p = &tile[k0][n0]; tile stored [K][N] row-major with row stride ld.
__device__ __forceinline__ v16h load_fragB(const _Float16* p, int ld, int lane) {
  const int h = (lane >> 4) & 1, n = lane & 15;
  v16h f;
#pragma unroll
  for (int i = 0; i < 8; ++i) { f[i] = p[(h * 8 + i) * ld + n]; f[i + 8] = p[(16 + h * 8 + i) * ld + n]; }
  return f;
}

// Async b128 copy of `n128` 16-byte packets from global f16 to LDS f16.
__device__ __forceinline__ void async_copy_b128(const _Float16* gsrc, _Float16* ldst,
                                                int tid, int n128) {
#if HAVE_ASYNC_LDS
  const AS1 char* g = (const AS1 char*)gsrc;
  AS3 char* l = (AS3 char*)ldst;
  for (int i = tid; i < n128; i += 256)
    __builtin_amdgcn_global_load_async_to_lds_b128(
        (AS1 v4i*)(g + (size_t)i * 16), (AS3 v4i*)(l + (size_t)i * 16), 0, 0);
  __builtin_amdgcn_s_wait_asynccnt(0);
#else
  for (int i = tid; i < n128; i += 256)
    ((v8h*)ldst)[i] = *(const v8h*)(gsrc + (size_t)i * 8);
#endif
}

// ---------------------------------------------------------------------------
// Kernel 1/4: C(M,N) = A(M,K) @ W(N,K)^T, fp32 inputs, f16 or f32 output.
// Block tile 128x128x32, 8 waves, each wave: 2x4 grid of 16x16 WMMA tiles.
// ---------------------------------------------------------------------------
#define GT_M 128
#define GT_N 128
#define GT_K 32

__global__ __launch_bounds__(256) void gemm_nt_kernel(
    const float* __restrict__ A, const float* __restrict__ Bw,
    void* __restrict__ Cout, int M, int N, int K, int outF16) {
  __shared__ _Float16 As[GT_M][GT_K];   // [M][K] row-major (A frags)
  __shared__ _Float16 Bs[GT_K][GT_N];   // [K][N] row-major (B frags)
  const int tid = threadIdx.x, lane = tid & 31, wave = tid >> 5;
  const int cl = lane & 15, mb = (lane >> 4) * 8;
  const int m0 = blockIdx.x * GT_M, n0 = blockIdx.y * GT_N;
  const int wm = (wave & 3) * 32;   // 4 waves along M
  const int wn = (wave >> 2) * 64;  // 2 waves along N

  v8f acc[2][4] = {};

  for (int k0 = 0; k0 < K; k0 += GT_K) {
#pragma unroll
    for (int i = 0; i < 4; ++i) {
      int id = tid + i * 256;           // 1024 float4 slots = 128 rows x 8
      int r = id >> 3, c4 = (id & 7) * 4;
      v4f a = *(const v4f*)(A + (size_t)(m0 + r) * K + k0 + c4);
      v4f w = *(const v4f*)(Bw + (size_t)(n0 + r) * K + k0 + c4);
#pragma unroll
      for (int j = 0; j < 4; ++j) {
        As[r][c4 + j] = (_Float16)a[j];
        Bs[c4 + j][r] = (_Float16)w[j];   // transpose into [K][N]
      }
    }
    if (k0 + GT_K < K) {
      __builtin_prefetch(A + (size_t)(m0 + (tid >> 3)) * K + k0 + GT_K + (tid & 7) * 4, 0, 1);
      __builtin_prefetch(Bw + (size_t)(n0 + (tid >> 3)) * K + k0 + GT_K + (tid & 7) * 4, 0, 1);
    }
    __syncthreads();
    v16h af[2], bf[4];
#pragma unroll
    for (int mi = 0; mi < 2; ++mi)
      af[mi] = load_fragA(&As[wm + mi * 16 + cl][0], 0, lane);
#pragma unroll
    for (int ni = 0; ni < 4; ++ni)
      bf[ni] = load_fragB(&Bs[0][wn + ni * 16], GT_N, lane);
#pragma unroll
    for (int mi = 0; mi < 2; ++mi)
#pragma unroll
      for (int ni = 0; ni < 4; ++ni)
        acc[mi][ni] = wmma_f16(af[mi], bf[ni], acc[mi][ni]);
    __syncthreads();
  }

#pragma unroll
  for (int mi = 0; mi < 2; ++mi)
#pragma unroll
    for (int ni = 0; ni < 4; ++ni) {
      int mg = m0 + wm + mi * 16 + mb;
      int ng = n0 + wn + ni * 16 + cl;
      if (outF16) {
        _Float16* C = (_Float16*)Cout;
#pragma unroll
        for (int r = 0; r < 8; ++r)
          C[(size_t)(mg + r) * N + ng] = (_Float16)acc[mi][ni][r];
      } else {
        float* C = (float*)Cout;
#pragma unroll
        for (int r = 0; r < 8; ++r)
          C[(size_t)(mg + r) * N + ng] = acc[mi][ni][r];
      }
    }
}

// ---------------------------------------------------------------------------
// Kernel 2: per-segment softmax attention (flash style).
// grid = (4 row-blocks, H=16, B*n_seg=32); block = 256 (8 waves x 16 rows).
// Layout: q/k/v f16 workspace is contiguously [B][n_seg][H][SEG][64].
// ---------------------------------------------------------------------------
__global__ __launch_bounds__(256) void attn_dot_kernel(
    const _Float16* __restrict__ qb, const _Float16* __restrict__ kb,
    const _Float16* __restrict__ vb, float* __restrict__ attb) {
  __shared__ _Float16 KT[64][128];     // K^T chunk: [dk][key]   (B frags of S)
  __shared__ _Float16 Vs[128][64];     // V chunk:   [key][dv]   (B frags of PV)
  __shared__ _Float16 Ps[8][16][32];   // wave-private P tiles   (A frags of PV)
  const int tid = threadIdx.x, lane = tid & 31, wave = tid >> 5;
  const int cl = lane & 15, mb = (lane >> 4) * 8;
  const size_t segOff = (size_t)blockIdx.z * 524288 + (size_t)blockIdx.y * 32768;
  const _Float16* qseg = qb + segOff;
  const _Float16* kseg = kb + segOff;
  const _Float16* vseg = vb + segOff;
  float* attseg = attb + segOff;
  const int qr0 = blockIdx.x * 128 + wave * 16;

  // Q A-fragments straight from global, prescaled by 1/sqrt(64)=0.125 (exact in f16)
  v16h aq[2];
#pragma unroll
  for (int c = 0; c < 2; ++c) {
    aq[c] = load_fragA(qseg + (size_t)(qr0 + cl) * 64, c * 32, lane);
#pragma unroll
    for (int i = 0; i < 16; ++i) aq[c][i] = aq[c][i] * (_Float16)0.125f;
  }

  float mrow[8], lrow[8];
#pragma unroll
  for (int r = 0; r < 8; ++r) { mrow[r] = -1e30f; lrow[r] = 0.f; }
  v8f o[4] = {};

  for (int ch = 0; ch < 4; ++ch) {          // 4 chunks of 128 keys
    __syncthreads();
    // V chunk: straight f16 copy -> async DMA to LDS (no VGPR round trip)
    async_copy_b128(vseg + (size_t)ch * 8192, &Vs[0][0], tid, 1024);
#pragma unroll
    for (int i = 0; i < 32; ++i) {          // K transposed into LDS
      int id = tid + i * 256;
      int s = id >> 6, d = id & 63;
      KT[d][s] = kseg[(size_t)ch * 8192 + id];
    }
    __syncthreads();

    for (int j = 0; j < 4; ++j) {           // 32-key blocks within chunk
      v8f sblk[2];
#pragma unroll
      for (int kt = 0; kt < 2; ++kt) {
        int key0 = j * 32 + kt * 16;
        v8f sb = {};
        sb = wmma_f16(aq[0], load_fragB(&KT[0][key0], 128, lane), sb);
        sb = wmma_f16(aq[1], load_fragB(&KT[32][key0], 128, lane), sb);
        sblk[kt] = sb;
      }
#pragma unroll
      for (int r = 0; r < 8; ++r) {
        float v = fmaxf(sblk[0][r], sblk[1][r]);
        v = fmaxf(v, __shfl_xor(v, 1, 32));
        v = fmaxf(v, __shfl_xor(v, 2, 32));
        v = fmaxf(v, __shfl_xor(v, 4, 32));
        v = fmaxf(v, __shfl_xor(v, 8, 32));
        float mn = fmaxf(mrow[r], v);
        float alpha = __expf(mrow[r] - mn);
        mrow[r] = mn;
        lrow[r] *= alpha;
#pragma unroll
        for (int ni = 0; ni < 4; ++ni) o[ni][r] *= alpha;
#pragma unroll
        for (int kt = 0; kt < 2; ++kt) {
          float p = __expf(sblk[kt][r] - mn);
          float ps = p;
          ps += __shfl_xor(ps, 1, 32);
          ps += __shfl_xor(ps, 2, 32);
          ps += __shfl_xor(ps, 4, 32);
          ps += __shfl_xor(ps, 8, 32);
          lrow[r] += ps;
          Ps[wave][mb + r][kt * 16 + cl] = (_Float16)p;   // wave-local, LDS in-order
        }
      }
      v16h pf = load_fragA(&Ps[wave][cl][0], 0, lane);
#pragma unroll
      for (int ni = 0; ni < 4; ++ni)
        o[ni] = wmma_f16(pf, load_fragB(&Vs[j * 32][ni * 16], 64, lane), o[ni]);
    }
  }

#pragma unroll
  for (int r = 0; r < 8; ++r) {
    float inv = 1.f / lrow[r];
#pragma unroll
    for (int ni = 0; ni < 4; ++ni)
      attseg[(size_t)(qr0 + mb + r) * 64 + ni * 16 + cl] = o[ni][r] * inv;
  }
}

// ---------------------------------------------------------------------------
// Kernel 3: recurrent compressive memory. 32 blocks = (b,h); scans 16 segments.
// mem (64x64) lives in WMMA accumulators (2 tiles/wave) across segments;
// denom[s,v] = rowsum(sq[s,:]) * Z[v]  (z's rows are identical broadcasts).
// ---------------------------------------------------------------------------
__global__ __launch_bounds__(256) void recurrent_kernel(
    const _Float16* __restrict__ qb, const _Float16* __restrict__ kb,
    const _Float16* __restrict__ vb, const float* __restrict__ betas,
    float* __restrict__ attb) {
  __shared__ _Float16 skT[64][128];    // sk^T chunk [dk][s]  (A frags, mem update)
  __shared__ _Float16 Vs2[128][64];    // v chunk [s][dv]     (B frags, mem update)
  __shared__ _Float16 sqs[128][64];    // sq chunk [s][dk]    (A frags, num GEMM)
  __shared__ _Float16 memL[64][64];    // mem [dk][dv] f16    (B frags, num GEMM)
  __shared__ float Zrun[64], Zseg[64], rs[512], gateL[64];
  const int tid = threadIdx.x, lane = tid & 31, wave = tid >> 5;
  const int cl = lane & 15, mb = (lane >> 4) * 8;
  const int b = blockIdx.x >> 4, h = blockIdx.x & 15;

  if (tid < 64) {
    Zrun[tid] = 0.f;
    gateL[tid] = 1.f / (1.f + __expf(-betas[h * 64 + tid]));
  }
  for (int i = tid; i < 64 * 64; i += 256) ((_Float16*)memL)[i] = (_Float16)0.f;
  v8f memAcc[2] = {};

  for (int n = 0; n < 16; ++n) {
    const size_t segOff = (size_t)(b * 16 + n) * 524288 + (size_t)h * 32768;
    const _Float16* qseg = qb + segOff;
    const _Float16* kseg = kb + segOff;
    const _Float16* vseg = vb + segOff;
    float* attseg = attb + segOff;
    __syncthreads();
    if (tid < 64) Zseg[tid] = 0.f;
    __syncthreads();
    {   // pass 1: Z (column sums of sk) and rs (row sums of sq)
      int d = tid & 63, sb = tid >> 6;
      float zs = 0.f;
      for (int s = sb * 128; s < sb * 128 + 128; ++s) {
        float kv = (float)kseg[(size_t)s * 64 + d];
        zs += (kv > 0.f) ? (kv + 1.f) : __expf(kv);      // elu(x)+1
      }
      atomicAdd(&Zseg[d], zs);
    }
#pragma unroll
    for (int ii = 0; ii < 2; ++ii) {
      int s = tid + ii * 256;
      float rsum = 0.f;
      for (int d = 0; d < 64; ++d) {
        float qv = (float)qseg[(size_t)s * 64 + d];
        rsum += (qv > 0.f) ? (qv + 1.f) : __expf(qv);
      }
      rs[s] = rsum;
    }
    __syncthreads();
    if (tid < 64) Zrun[tid] += Zseg[tid];   // z updated BEFORE att_mem (ref order)
    __syncthreads();

    for (int ch = 0; ch < 4; ++ch) {        // chunks of 128 rows
      const int s0 = ch * 128;
      // V chunk: async DMA straight copy into LDS
      async_copy_b128(vseg + (size_t)s0 * 64, &Vs2[0][0], tid, 1024);
      {
        int d = tid & 63, sb = tid >> 6;
        for (int si = sb * 32; si < sb * 32 + 32; ++si) {
          int s = s0 + si;
          float kv = (float)kseg[(size_t)s * 64 + d];
          skT[d][si] = (_Float16)((kv > 0.f) ? (kv + 1.f) : __expf(kv));
          float qv = (float)qseg[(size_t)s * 64 + d];
          sqs[si][d] = (_Float16)((qv > 0.f) ? (qv + 1.f) : __expf(qv));
        }
      }
      __syncthreads();
      // att_mem = (sq@mem)/denom; gate-combine with att_dot in place
      for (int t = wave; t < 32; t += 8) {
        int mt = t >> 2, nt = t & 3;
        const _Float16* arow = &sqs[mt * 16 + cl][0];
        v8f num = {};
        num = wmma_f16(load_fragA(arow, 0, lane), load_fragB(&memL[0][nt * 16], 64, lane), num);
        num = wmma_f16(load_fragA(arow, 32, lane), load_fragB(&memL[32][nt * 16], 64, lane), num);
#pragma unroll
        for (int r = 0; r < 8; ++r) {
          int s = s0 + mt * 16 + mb + r;
          int v = nt * 16 + cl;
          float am = num[r] / (rs[s] * Zrun[v]);
          float g = gateL[v];
          size_t idx = (size_t)s * 64 + v;
          attseg[idx] = g * am + (1.f - g) * attseg[idx];
        }
      }
      // mem += sk^T @ v (partial over this chunk), mem uses OLD value above
#pragma unroll
      for (int u = 0; u < 2; ++u) {
        int t = wave * 2 + u, mt = t >> 2, nt = t & 3;
#pragma unroll
        for (int ks = 0; ks < 128; ks += 32)
          memAcc[u] = wmma_f16(load_fragA(&skT[mt * 16 + cl][ks], 0, lane),
                               load_fragB(&Vs2[ks][nt * 16], 64, lane), memAcc[u]);
      }
      __syncthreads();
    }
    // publish updated mem (f16) for next segment's num GEMM
#pragma unroll
    for (int u = 0; u < 2; ++u) {
      int t = wave * 2 + u, mt = t >> 2, nt = t & 3;
#pragma unroll
      for (int r = 0; r < 8; ++r)
        memL[mt * 16 + mb + r][nt * 16 + cl] = (_Float16)memAcc[u][r];
    }
    __syncthreads();
  }
}

// ---------------------------------------------------------------------------
// Launch: 3 projection GEMMs (f16 out) -> segment softmax -> recurrent scan
//         -> output GEMM (f32 out).  Workspace: q,k,v f16 (3x32MB) + att f32 (64MB).
// ---------------------------------------------------------------------------
extern "C" void kernel_launch(void* const* d_in, const int* in_sizes, int n_in,
                              void* d_out, int out_size, void* d_ws, size_t ws_size,
                              hipStream_t stream) {
  const float* x    = (const float*)d_in[0];
  const float* Wq   = (const float*)d_in[1];
  const float* Wk   = (const float*)d_in[2];
  const float* Wv   = (const float*)d_in[3];
  const float* Wout = (const float*)d_in[4];
  const float* beta = (const float*)d_in[5];

  const int M = 16384, N = 1024, K = 1024;   // B*T rows
  _Float16* qbuf = (_Float16*)d_ws;
  _Float16* kbuf = qbuf + (size_t)M * N;
  _Float16* vbuf = kbuf + (size_t)M * N;
  float*    attb = (float*)(vbuf + (size_t)M * N);
  float*    out  = (float*)d_out;

  dim3 gg(M / GT_M, N / GT_N);
  gemm_nt_kernel<<<gg, 256, 0, stream>>>(x, Wq, qbuf, M, N, K, 1);
  gemm_nt_kernel<<<gg, 256, 0, stream>>>(x, Wk, kbuf, M, N, K, 1);
  gemm_nt_kernel<<<gg, 256, 0, stream>>>(x, Wv, vbuf, M, N, K, 1);

  dim3 ga(4, 16, 32);   // row-blocks, H, B*n_seg
  attn_dot_kernel<<<ga, 256, 0, stream>>>(qbuf, kbuf, vbuf, attb);

  recurrent_kernel<<<32, 256, 0, stream>>>(qbuf, kbuf, vbuf, beta, attb);

  gemm_nt_kernel<<<gg, 256, 0, stream>>>(attb, Wout, out, M, N, K, 0);
}
